// MultiHead_attn_50818053046565
// MI455X (gfx1250) — matrix operand order repaired and checked by reading, not compiled, not run on hardware
//
#include <hip/hip_runtime.h>

// MultiHead attention (B=2, S=2048, D=1024, H=16, dh=64) for MI455X / gfx1250.
// Pipeline:
//   [cvt_x_f16]  x(f32) -> xh(f16)                      (bandwidth prepass)
//   [cvt_w_t]    w[z](f32,[K][N]) -> wht[z](f16,[N][K]) (convert + transpose)
//   [qkv_gemm]   async-LDS double-buffered f16 WMMA GEMM -> Q,K,V (f16, ws)
//   [flash_attn] f16 WMMA flash attention + residual -> d_out (f32)
//   [residual_layernorm] in-place row LayerNorm on d_out

#define DMODEL 1024
#define SEQ    2048
#define BATCH  2
#define NHEAD  16
#define DHEAD  64
#define MROWS  (BATCH * SEQ)   // 4096
#define KSTEPS (DMODEL / 32)   // 32

typedef _Float16 h8   __attribute__((ext_vector_type(8)));
typedef _Float16 v16h __attribute__((ext_vector_type(16)));
typedef float    v8f  __attribute__((ext_vector_type(8)));
typedef float    f4   __attribute__((ext_vector_type(4)));

__device__ __forceinline__ v8f zero8() {
    v8f z;
    #pragma unroll
    for (int i = 0; i < 8; ++i) z[i] = 0.0f;
    return z;
}

// Build a 16-element fragment from two contiguous 8-half runs.
__device__ __forceinline__ v16h make_frag(h8 lo, h8 hi) {
    v16h f;
    #pragma unroll
    for (int i = 0; i < 8; ++i) { f[i] = lo[i]; f[8 + i] = hi[i]; }
    return f;
}

// CDNA5 async copy: 16B global -> LDS, tracked by ASYNCcnt (no VGPR data path).
__device__ __forceinline__ void async_ld_b128(uint32_t lds_addr, const _Float16* g) {
    asm volatile("global_load_async_to_lds_b128 %0, %1, off"
                 :: "v"(lds_addr), "v"(g) : "memory");
}
__device__ __forceinline__ void wait_async0() {
    asm volatile("s_wait_asynccnt 0x0" ::: "memory");
}
__device__ __forceinline__ uint32_t lds_addr_of(const void* p) {
    return (uint32_t)(uintptr_t)p;
}

// ---------------------------------------------------------------------------
// Prepass 1: x (f32) -> xh (f16).  8 elements/thread, fully coalesced.
// ---------------------------------------------------------------------------
__global__ __launch_bounds__(256) void cvt_x_f16(const float* __restrict__ x,
                                                 _Float16* __restrict__ xh)
{
    const size_t i = (size_t)blockIdx.x * 256 + threadIdx.x;
    const f4* src = (const f4*)x;
    f4 a = src[2 * i], b = src[2 * i + 1];
    h8 o;
    #pragma unroll
    for (int e = 0; e < 4; ++e) { o[e] = (_Float16)a[e]; o[4 + e] = (_Float16)b[e]; }
    ((h8*)xh)[i] = o;
}

// ---------------------------------------------------------------------------
// Prepass 2: wht[z][n][k] = f16( w[z][k][n] )  (tiled LDS transpose)
// grid = (N/32, K/32, 3), block = 256
// ---------------------------------------------------------------------------
__global__ __launch_bounds__(256) void cvt_w_t(
    const float* __restrict__ w1, const float* __restrict__ w2, const float* __restrict__ w3,
    _Float16* __restrict__ wht)
{
    const int z = blockIdx.z;
    const float* w = (z == 0) ? w1 : ((z == 1) ? w2 : w3);
    _Float16* out = wht + (size_t)z * DMODEL * DMODEL;

    __shared__ _Float16 tile[32][33];
    const int n0 = blockIdx.x * 32, k0 = blockIdx.y * 32;
    const int tx = threadIdx.x & 31;
    const int ty = threadIdx.x >> 5;
    #pragma unroll
    for (int j = 0; j < 4; ++j)
        tile[ty + 8 * j][tx] = (_Float16)w[(size_t)(k0 + ty + 8 * j) * DMODEL + n0 + tx];
    __syncthreads();
    #pragma unroll
    for (int j = 0; j < 4; ++j)
        out[(size_t)(n0 + ty + 8 * j) * DMODEL + k0 + tx] = tile[tx][ty + 8 * j];
}

// ---------------------------------------------------------------------------
// Kernel 1: QKV projection, all-f16 operands, async double-buffered LDS.
// grid = (N/128, M/128, 3), block = 256 (8 waves, wave tile 64x32)
// ---------------------------------------------------------------------------
__global__ __launch_bounds__(256) void qkv_gemm(
    const _Float16* __restrict__ xh,     // [4096][1024] f16
    const _Float16* __restrict__ wht,    // [3][1024(N)][1024(K)] f16
    const float* __restrict__ b1, const float* __restrict__ b2, const float* __restrict__ b3,
    _Float16* __restrict__ qkv)
{
    const int z = blockIdx.z;
    const _Float16* w_t  = wht + (size_t)z * DMODEL * DMODEL;
    const float*    bias = (z == 0) ? b1 : ((z == 1) ? b2 : b3);
    _Float16* out = qkv + (size_t)z * ((size_t)MROWS * DMODEL);

    const int n0   = blockIdx.x * 128;
    const int m0   = blockIdx.y * 128;
    const int t    = threadIdx.x;
    const int lane = t & 31;
    const int wave = t >> 5;
    const int wm   = wave & 1;    // 64-row strip
    const int wn   = wave >> 1;   // 32-col strip
    const int half = lane >> 4;
    const int l16  = lane & 15;

    // +8 half pad -> 20-dword row stride, conflict-free 16-lane b128 reads
    __shared__ alignas(16) _Float16 As[2][128][40];  // [m][k]
    __shared__ alignas(16) _Float16 Bs[2][128][40];  // [n][k] (w pre-transposed)

    v8f acc[4][2];
    #pragma unroll
    for (int i = 0; i < 4; ++i)
        #pragma unroll
        for (int j = 0; j < 2; ++j) acc[i][j] = zero8();

    // stage one 128x32 f16 tile pair via async DMA (4 b128 ops / thread)
    auto stage = [&](int buf, int k0) {
        #pragma unroll
        for (int j = 0; j < 2; ++j) {
            const int ch  = t + 256 * j;   // 0..511
            const int row = ch >> 2;
            const int cc  = (ch & 3) * 8;  // half offset within row
            async_ld_b128(lds_addr_of(&As[buf][row][cc]),
                          xh + (size_t)(m0 + row) * DMODEL + k0 + cc);
            async_ld_b128(lds_addr_of(&Bs[buf][row][cc]),
                          w_t + (size_t)(n0 + row) * DMODEL + k0 + cc);
        }
    };

    stage(0, 0);
    for (int it = 0; it < KSTEPS; ++it) {
        wait_async0();        // this wave's copies for tile `it` landed in LDS
        __syncthreads();      // everyone's landed; prev compute done -> other buf free
        if (it + 1 < KSTEPS) stage((it + 1) & 1, (it + 1) * 32);  // overlap with WMMAs
        const int buf = it & 1;

        v16h a[4];
        #pragma unroll
        for (int fm = 0; fm < 4; ++fm) {
            const h8* p8 = (const h8*)&As[buf][wm * 64 + fm * 16 + l16][0];
            a[fm] = make_frag(p8[half], p8[2 + half]);    // A: K = {0..7,16..23}/{8..15,24..31}
        }
        v16h bf[2];
        #pragma unroll
        for (int fn = 0; fn < 2; ++fn) {
            const h8* p8 = (const h8*)&Bs[buf][wn * 32 + fn * 16 + l16][0];
            bf[fn] = make_frag(p8[half], p8[2 + half]);  // B: K = {0..7,16..23}/{8..15,24..31}
        }
        #pragma unroll
        for (int fm = 0; fm < 4; ++fm)
            #pragma unroll
            for (int fn = 0; fn < 2; ++fn)
                acc[fm][fn] = __builtin_amdgcn_wmma_f32_16x16x32_f16(
                    false, a[fm], false, bf[fn], (short)0, acc[fm][fn], false, false);
    }

    // epilogue: bias + f16 store
    #pragma unroll
    for (int fn = 0; fn < 2; ++fn) {
        const int nn = n0 + wn * 32 + fn * 16 + l16;
        const float bv = bias[nn];
        #pragma unroll
        for (int fm = 0; fm < 4; ++fm) {
            #pragma unroll
            for (int r = 0; r < 8; ++r) {
                const int mm = m0 + wm * 64 + fm * 16 + r + 8 * half;
                out[(size_t)mm * DMODEL + nn] = (_Float16)(acc[fm][fn][r] + bv);
            }
        }
    }
}

// ---------------------------------------------------------------------------
// Kernel 2: flash attention + residual.  grid = (S/64, H, B), block = 128.
// K tile staged via async DMA; V transposed through registers (needs shuffle).
// ---------------------------------------------------------------------------
__global__ __launch_bounds__(128) void flash_attn(
    const _Float16* __restrict__ qkv,
    const float* __restrict__ x,
    float* __restrict__ out)
{
    const int qb   = blockIdx.x;
    const int h    = blockIdx.y;
    const int b    = blockIdx.z;
    const int t    = threadIdx.x;
    const int lane = t & 31;
    const int wave = t >> 5;
    const int half = lane >> 4;
    const int l16  = lane & 15;

    const _Float16* Q = qkv;
    const _Float16* K = qkv + (size_t)MROWS * DMODEL;
    const _Float16* V = qkv + 2u * (size_t)MROWS * DMODEL;
    const size_t headbase = (size_t)(b * SEQ) * DMODEL + (size_t)h * DHEAD;

    __shared__ alignas(16) _Float16 Ks[64][72];      // [kv][d]  (async-staged)
    __shared__ alignas(16) _Float16 Vt[64][72];      // [d][kv]  (transposed at stage)
    __shared__ alignas(16) _Float16 Ps[4][16][72];   // per-wave P bridge (C->A layout)

    v16h qf[2];
    #pragma unroll
    for (int c = 0; c < 2; ++c) {
        const h8* p8 = (const h8*)(Q + headbase +
                       (size_t)(qb * 64 + wave * 16 + l16) * DMODEL + 32 * c);
        qf[c] = make_frag(p8[half], p8[2 + half]);
    }

    v8f o[4];
    float mrun[8], lrun[8];
    #pragma unroll
    for (int i = 0; i < 4; ++i) o[i] = zero8();
    #pragma unroll
    for (int r = 0; r < 8; ++r) { mrun[r] = -1e30f; lrun[r] = 0.0f; }

    for (int kv0 = 0; kv0 < SEQ; kv0 += 64) {
        // ---- stage K via async DMA: 64x64 f16 = 512 chunks, 4/thread ----
        #pragma unroll
        for (int j = 0; j < 4; ++j) {
            const int ch  = t + 128 * j;   // 0..511
            const int r   = ch >> 3;
            const int cc  = (ch & 7) * 8;
            async_ld_b128(lds_addr_of(&Ks[r][cc]),
                          K + headbase + (size_t)(kv0 + r) * DMODEL + cc);
        }
        // ---- stage V transposed (vector global load, scalar LDS scatter) ----
        {
            const int r  = t >> 1;
            const int c0 = (t & 1) * 32;
            const h8* vp = (const h8*)(V + headbase + (size_t)(kv0 + r) * DMODEL + c0);
            #pragma unroll
            for (int j = 0; j < 4; ++j) {
                h8 v = vp[j];
                #pragma unroll
                for (int e = 0; e < 8; ++e) Vt[c0 + 8 * j + e][r] = v[e];
            }
        }
        wait_async0();
        __syncthreads();

        // ---- scores: S = Q @ K^T ----
        v8f s[4];
        #pragma unroll
        for (int fn = 0; fn < 4; ++fn) {
            s[fn] = zero8();
            #pragma unroll
            for (int c = 0; c < 2; ++c) {
                const h8* p8 = (const h8*)&Ks[fn * 16 + l16][32 * c];
                v16h kb = make_frag(p8[half], p8[2 + half]);
                s[fn] = __builtin_amdgcn_wmma_f32_16x16x32_f16(
                    false, qf[c], false, kb, (short)0, s[fn], false, false);
            }
        }
        // ---- online softmax ----
        float mnew[8], alpha[8];
        #pragma unroll
        for (int r = 0; r < 8; ++r) {
            #pragma unroll
            for (int fn = 0; fn < 4; ++fn) s[fn][r] *= 0.125f;  // 1/sqrt(dh)
            float mx = fmaxf(fmaxf(s[0][r], s[1][r]), fmaxf(s[2][r], s[3][r]));
            #pragma unroll
            for (int off = 8; off >= 1; off >>= 1) mx = fmaxf(mx, __shfl_xor(mx, off, 16));
            mnew[r]  = fmaxf(mrun[r], mx);
            alpha[r] = __expf(mrun[r] - mnew[r]);
            mrun[r]  = mnew[r];
        }
        #pragma unroll
        for (int r = 0; r < 8; ++r) {
            float sum = 0.0f;
            #pragma unroll
            for (int fn = 0; fn < 4; ++fn) {
                float p = __expf(s[fn][r] - mnew[r]);
                sum += p;
                Ps[wave][r + 8 * half][fn * 16 + l16] = (_Float16)p;
            }
            #pragma unroll
            for (int off = 8; off >= 1; off >>= 1) sum += __shfl_xor(sum, off, 16);
            lrun[r] = lrun[r] * alpha[r] + sum;
            #pragma unroll
            for (int od = 0; od < 4; ++od) o[od][r] *= alpha[r];
        }
        // ---- O += P @ V ----
        #pragma unroll
        for (int kc = 0; kc < 2; ++kc) {
            const h8* p8 = (const h8*)&Ps[wave][l16][32 * kc];
            v16h pa = make_frag(p8[half], p8[2 + half]);
            #pragma unroll
            for (int od = 0; od < 4; ++od) {
                const h8* q8 = (const h8*)&Vt[od * 16 + l16][32 * kc];
                v16h vb = make_frag(q8[half], q8[2 + half]);
                o[od] = __builtin_amdgcn_wmma_f32_16x16x32_f16(
                    false, pa, false, vb, (short)0, o[od], false, false);
            }
        }
        __syncthreads();
    }

    // ---- epilogue: normalize, residual add, fp32 store ----
    #pragma unroll
    for (int r = 0; r < 8; ++r) {
        const float inv = 1.0f / lrun[r];
        const int qr = qb * 64 + wave * 16 + r + 8 * half;
        const size_t base = (size_t)(b * SEQ + qr) * DMODEL + (size_t)h * DHEAD;
        #pragma unroll
        for (int od = 0; od < 4; ++od) {
            const int d = od * 16 + l16;
            out[base + d] = o[od][r] * inv + x[base + d];
        }
    }
}

// ---------------------------------------------------------------------------
// Kernel 3: in-place LayerNorm over D=1024 per row (gamma=1, beta=0, eps=1e-3)
// ---------------------------------------------------------------------------
__global__ __launch_bounds__(256) void residual_layernorm(float* __restrict__ io)
{
    const int t = threadIdx.x;
    f4* p = (f4*)(io + (size_t)blockIdx.x * DMODEL);
    f4 v = p[t];

    __shared__ float red[8];
    float s = v[0] + v[1] + v[2] + v[3];
    #pragma unroll
    for (int off = 16; off >= 1; off >>= 1) s += __shfl_xor(s, off, 32);
    if ((t & 31) == 0) red[t >> 5] = s;
    __syncthreads();
    float tot = 0.0f;
    #pragma unroll
    for (int i = 0; i < 8; ++i) tot += red[i];
    const float mean = tot * (1.0f / DMODEL);
    __syncthreads();

    f4 d;
    #pragma unroll
    for (int j = 0; j < 4; ++j) d[j] = v[j] - mean;
    float vs = d[0] * d[0] + d[1] * d[1] + d[2] * d[2] + d[3] * d[3];
    #pragma unroll
    for (int off = 16; off >= 1; off >>= 1) vs += __shfl_xor(vs, off, 32);
    if ((t & 31) == 0) red[t >> 5] = vs;
    __syncthreads();
    float vtot = 0.0f;
    #pragma unroll
    for (int i = 0; i < 8; ++i) vtot += red[i];
    const float inv = rsqrtf(vtot * (1.0f / DMODEL) + 1e-3f);

    f4 outv;
    #pragma unroll
    for (int j = 0; j < 4; ++j) outv[j] = d[j] * inv;
    p[t] = outv;
}

// ---------------------------------------------------------------------------
extern "C" void kernel_launch(void* const* d_in, const int* in_sizes, int n_in,
                              void* d_out, int out_size, void* d_ws, size_t ws_size,
                              hipStream_t stream)
{
    (void)in_sizes; (void)n_in; (void)out_size; (void)ws_size;
    const float* x  = (const float*)d_in[0];
    const float* w1 = (const float*)d_in[1];
    const float* w2 = (const float*)d_in[2];
    const float* w3 = (const float*)d_in[3];
    const float* b1 = (const float*)d_in[4];
    const float* b2 = (const float*)d_in[5];
    const float* b3 = (const float*)d_in[6];
    float* out = (float*)d_out;

    // workspace layout (38 MB):
    //   [0,   24MB)  qkv  : 3 x [4096][1024] f16
    //   [24MB,32MB)  xh   : [4096][1024] f16
    //   [32MB,38MB)  wht  : 3 x [1024][1024] f16 (transposed)
    char* ws = (char*)d_ws;
    _Float16* qkv = (_Float16*)(ws);
    _Float16* xh  = (_Float16*)(ws + (size_t)24 * 1024 * 1024);
    _Float16* wht = (_Float16*)(ws + (size_t)32 * 1024 * 1024);

    cvt_x_f16<<<(MROWS * DMODEL) / (256 * 8), 256, 0, stream>>>(x, xh);
    dim3 gt(DMODEL / 32, DMODEL / 32, 3);
    cvt_w_t<<<gt, 256, 0, stream>>>(w1, w2, w3, wht);

    dim3 g1(DMODEL / 128, MROWS / 128, 3);
    qkv_gemm<<<g1, 256, 0, stream>>>(xh, wht, b1, b2, b3, qkv);

    dim3 g2(SEQ / 64, NHEAD, BATCH);
    flash_attn<<<g2, 128, 0, stream>>>(qkv, x, out);

    residual_layernorm<<<MROWS, 256, 0, stream>>>(out);
}
